// FraudMLPWithEmbedding_89481348645267
// MI455X (gfx1250) — matrix operand order repaired and checked by reading, not compiled, not run on hardware
//
#include <hip/hip_runtime.h>

// ---- CDNA5 (gfx1250) wave32 WMMA / TDM types -------------------------------
typedef __attribute__((ext_vector_type(16))) __bf16       v16bf;
typedef __attribute__((ext_vector_type(2)))  __bf16       v2bf;
typedef __attribute__((ext_vector_type(8)))  float        v8f;
typedef __attribute__((ext_vector_type(8)))  unsigned int u8v;
typedef __attribute__((ext_vector_type(4)))  unsigned int u4v;
typedef __attribute__((ext_vector_type(8)))  int          i8v;
typedef __attribute__((ext_vector_type(4)))  int          i4v;
typedef __attribute__((ext_vector_type(4)))  float        f4v;

#define B_TOTAL   262144
#define NUMF      16
#define IN_DIM    128
#define HID       256
#define ROWS_PB   128                    // rows per block (8 waves x 16 rows)
#define THREADS   256
#define W1_WORDS  (IN_DIM * HID / 2)     // packed bf16 pairs per array: 16384
#define W1_BYTES  (2 * W1_WORDS * 4)     // hi + lo image: 131072 bytes

// ---- f32 -> bf16 helpers ---------------------------------------------------
__device__ __forceinline__ unsigned short f2bf(float f) {
  unsigned int u = __builtin_bit_cast(unsigned int, f);
  u += 0x7FFFu + ((u >> 16) & 1u);
  return (unsigned short)(u >> 16);
}
// pack bf16(f0) | bf16(f1)<<16 — single v_cvt_pk_bf16_f32 when available
__device__ __forceinline__ unsigned int pack_bf16(float f0, float f1) {
#if __has_builtin(__builtin_amdgcn_cvt_pk_bf16_f32)
  v2bf p = __builtin_amdgcn_cvt_pk_bf16_f32(f0, f1);
  return __builtin_bit_cast(unsigned int, p);
#else
  return (unsigned int)f2bf(f0) | ((unsigned int)f2bf(f1) << 16);
#endif
}
// pack bf16 of the residuals (f - hi) given the packed hi word
__device__ __forceinline__ unsigned int pack_bf16_res(float f0, float f1,
                                                      unsigned int hw) {
  float r0 = f0 - __builtin_bit_cast(float, hw << 16);
  float r1 = f1 - __builtin_bit_cast(float, hw & 0xFFFF0000u);
  return pack_bf16(r0, r1);
}

// swizzled word index for the WMMA-B fragment layout:
// [nb(16)][kb(4)][lane(32)][8 words]; word j packs K=2j,2j+1 for lane's column
__device__ __forceinline__ int w1_swizzle(int n, int kp) {
  const int nb   = n >> 4;
  const int kb   = kp >> 4;
  const int lane = (n & 15) + (((kp >> 3) & 1) << 4);
  const int j    = kp & 7;
  return (((nb * 4 + kb) * 32 + lane) << 3) + j;
}

// ---- prep kernel: W1 f32 -> bf16 hi/lo, swizzled, ONCE into workspace ------
__global__ __launch_bounds__(THREADS)
void fraud_prep_w1(const float* __restrict__ W1, unsigned int* __restrict__ ws) {
  const int n = threadIdx.x;                 // hidden column 0..255
  #pragma unroll
  for (int kpi = 0; kpi < 4; ++kpi) {
    const int kp = blockIdx.x * 4 + kpi;     // 16 blocks x 4 = 64 K-pairs
    float f0 = W1[(2 * kp)     * HID + n];
    float f1 = W1[(2 * kp + 1) * HID + n];
    unsigned int hw = pack_bf16(f0, f1);
    unsigned int lw = pack_bf16_res(f0, f1, hw);
    const int widx = w1_swizzle(n, kp);
    ws[widx]            = hw;
    ws[W1_WORDS + widx] = lw;
  }
}

// ---- main kernel -----------------------------------------------------------
__global__ __launch_bounds__(THREADS)
void fraud_mlp_wmma(const float* __restrict__ x_num,
                    const int*   __restrict__ x_cat,
                    const float* __restrict__ emb0,
                    const float* __restrict__ emb1,
                    const float* __restrict__ emb2,
                    const float* __restrict__ W1,
                    const float* __restrict__ b1,
                    const float* __restrict__ W2,
                    const float* __restrict__ b2,
                    const unsigned int* __restrict__ wsW1,
                    int useWs,
                    float* __restrict__ out)
{
  // hi image at [0, W1_WORDS), lo image at [W1_WORDS, 2*W1_WORDS)  (128 KB)
  __shared__ unsigned int sW1[2 * W1_WORDS];

  const int t = threadIdx.x;

  if (useWs) {
#if __has_builtin(__builtin_amdgcn_tensor_load_to_lds)
    // --- Tensor Data Mover: one DMA brings the whole 128KB W1 image to LDS --
    if (t < 32) {                              // wave 0 issues the TDM op
      const unsigned long long ga =
          (unsigned long long)(const void*)wsW1;
      const unsigned int ldsOff =
          (unsigned int)(unsigned long long)(const void*)&sW1[0];
      const unsigned int TD0 = 2 * W1_WORDS;   // 32768 dwords, 1 row
      // D# group 0: count=1 | lds_addr | global_addr[56:0] | type=2
      u4v g0 = { 1u,
                 ldsOff,
                 (unsigned int)ga,
                 (unsigned int)((ga >> 32) & 0x01FFFFFFu) | 0x80000000u };
      // D# group 1: data_size=4B; tensor_dim0=tile_dim0=32768; tensor_dim1=1
      i8v g1 = { (int)(2u << 16),                         // data_size=2 (4B)
                 (int)((TD0 & 0xFFFFu) << 16),            // tensor_dim0[15:0]
                 (int)((TD0 >> 16) | (1u << 16)),         // td0[31:16]|td1[15:0]
                 (int)((TD0 & 0xFFFFu) << 16),            // tile_dim0
                 0,                                       // tile_dim1/2 unused
                 (int)TD0,                                // dim0 stride lo
                 0, 0 };
      i4v g2 = { 0, 0, 0, 0 };                            // <=2D: groups 2/3 off
      i4v g3 = { 0, 0, 0, 0 };
  #if __has_include(<hip/amd_detail/amd_gfx1250_TDM.h>)
      i8v g4 = { 0, 0, 0, 0, 0, 0, 0, 0 };                // clang-23 6-arg form
      __builtin_amdgcn_tensor_load_to_lds(g0, g1, g2, g3, g4, 0);
  #else
      __builtin_amdgcn_tensor_load_to_lds(g0, g1, g2, g3, 0);
  #endif
  #if __has_builtin(__builtin_amdgcn_s_wait_tensorcnt)
      __builtin_amdgcn_s_wait_tensorcnt(0);
  #else
      asm volatile("s_wait_tensorcnt 0x0" ::: "memory");
  #endif
    }
#else
    // --- fallback: coalesced b128 copy of the prebuilt image ---------------
    #pragma unroll
    for (int i = 0; i < 32; ++i) {
      const int w = (i * THREADS + t) * 4;
      *(u4v*)&sW1[w] = *(const u4v*)&wsW1[w];
    }
#endif
  } else {
    // --- no-workspace fallback: convert + swizzle in-kernel ----------------
    const int n = t;
    for (int kp = 0; kp < IN_DIM / 2; ++kp) {
      float f0 = W1[(2 * kp)     * HID + n];
      float f1 = W1[(2 * kp + 1) * HID + n];
      unsigned int hw = pack_bf16(f0, f1);
      unsigned int lw = pack_bf16_res(f0, f1, hw);
      const int widx = w1_swizzle(n, kp);
      sW1[widx]            = hw;
      sW1[W1_WORDS + widx] = lw;
    }
  }
  __syncthreads();

  const int wid  = t >> 5;        // wave 0..7
  const int lane = t & 31;
  const int half = lane >> 4;     // lane group (K-half for A, M+8 for C/D)
  const int mrow = lane & 15;     // this lane's A row / C column
  const long rowBase = (long)blockIdx.x * ROWS_PB + (long)wid * 16;
  const long r = rowBase + mrow;

  // embedding gather indices for this lane's row (tables are L2-resident)
  const int i0 = x_cat[r * 3 + 0];
  const int i1 = x_cat[r * 3 + 1];
  const int i2 = x_cat[r * 3 + 2];
  const float* pnum = x_num + r * NUMF;
  const float* p0   = emb0 + (long)i0 * 64;
  const float* p1   = emb1 + (long)i1 * 32;
  const float* p2   = emb2 + (long)i2 * 16;

  // 4 consecutive features of the concatenated vector starting at k (k%4==0);
  // 8-aligned runs never cross segment boundaries (16 / 80 / 112).
  auto seg4 = [&](int k) -> f4v {
    const float* p;
    if (k < 16)        p = pnum + k;
    else if (k < 80)   p = p0 + (k - 16);
    else if (k < 112)  p = p1 + (k - 80);
    else               p = p2 + (k - 112);
    return *(const f4v*)p;
  };

  // ---- A fragments (16 rows x 128 K) resident in registers, bf16 hi/lo ----
  v16bf Ahi[4], Alo[4];
  #pragma unroll
  for (int kb = 0; kb < 4; ++kb) {
    // A k order used here (contiguous, matches w1_swizzle): lanes<16 hold K={0..15}, lanes>=16 {16..31}
    const int base0 = kb * 32 + (half ? 16 : 0);
    const int base1 = base0 + 8;
    f4v q0 = seg4(base0);
    f4v q1 = seg4(base0 + 4);
    f4v q2 = seg4(base1);
    f4v q3 = seg4(base1 + 4);
    float vals[16];
    #pragma unroll
    for (int e = 0; e < 4; ++e) {
      vals[e] = q0[e]; vals[4 + e] = q1[e]; vals[8 + e] = q2[e]; vals[12 + e] = q3[e];
    }
    u8v uh, ul;
    #pragma unroll
    for (int w = 0; w < 8; ++w) {
      float f0 = vals[2 * w], f1 = vals[2 * w + 1];
      unsigned int hw = pack_bf16(f0, f1);
      uh[w] = hw;
      ul[w] = pack_bf16_res(f0, f1, hw);
    }
    Ahi[kb] = __builtin_bit_cast(v16bf, uh);
    Alo[kb] = __builtin_bit_cast(v16bf, ul);
  }

  // ---- GEMM1 (WMMA, fp32-via-bf16 hi/lo) fused with bias+ReLU+GEMM2 -------
  float dot[8] = {0.f, 0.f, 0.f, 0.f, 0.f, 0.f, 0.f, 0.f};

  for (int nb = 0; nb < 16; ++nb) {
    v8f acc = {0.f, 0.f, 0.f, 0.f, 0.f, 0.f, 0.f, 0.f};
    #pragma unroll
    for (int kb = 0; kb < 4; ++kb) {
      const int fbase = ((nb * 4 + kb) * 32 + lane) << 3;
      u8v uhw = *(const u8v*)&sW1[fbase];             // 2x ds_load_b128
      u8v ulw = *(const u8v*)&sW1[W1_WORDS + fbase];  // 2x ds_load_b128
      v16bf Bhi = __builtin_bit_cast(v16bf, uhw);
      v16bf Blo = __builtin_bit_cast(v16bf, ulw);
      acc = __builtin_amdgcn_wmma_f32_16x16x32_bf16(false, Ahi[kb], false, Bhi,
                                                    (short)0, acc, false, false);
      acc = __builtin_amdgcn_wmma_f32_16x16x32_bf16(false, Alo[kb], false, Bhi,
                                                    (short)0, acc, false, false);
      acc = __builtin_amdgcn_wmma_f32_16x16x32_bf16(false, Ahi[kb], false, Blo,
                                                    (short)0, acc, false, false);
    }
    // C/D layout: VGPR j, lanes 0-15 -> row j, lanes 16-31 -> row j+8; col=lane%16
    const int   n    = nb * 16 + mrow;
    const float bias = b1[n];
    const float w2v  = W2[n];
    #pragma unroll
    for (int j = 0; j < 8; ++j) {
      float h = acc[j] + bias;
      h = h > 0.f ? h : 0.f;                 // ReLU
      dot[j] = fmaf(h, w2v, dot[j]);         // fused GEMM2 partial dot
    }
  }

  // ---- reduce GEMM2 partials across the 16 column-lanes (wave32 butterfly) -
  #pragma unroll
  for (int j = 0; j < 8; ++j) {
    float v = dot[j];
    v += __shfl_xor(v, 1, 32);
    v += __shfl_xor(v, 2, 32);
    v += __shfl_xor(v, 4, 32);
    v += __shfl_xor(v, 8, 32);
    dot[j] = v;
  }

  // lane 0 holds rows 0..7, lane 16 holds rows 8..15
  if (mrow == 0) {
    const float bb = b2[0];
    #pragma unroll
    for (int j = 0; j < 8; ++j) {
      float z = dot[j] + bb;
      out[rowBase + half * 8 + j] = 1.0f / (1.0f + __expf(-z));
    }
  }
}

extern "C" void kernel_launch(void* const* d_in, const int* in_sizes, int n_in,
                              void* d_out, int out_size, void* d_ws, size_t ws_size,
                              hipStream_t stream) {
  (void)in_sizes; (void)n_in; (void)out_size;
  const float* x_num = (const float*)d_in[0];
  const int*   x_cat = (const int*)  d_in[1];
  const float* emb0  = (const float*)d_in[2];
  const float* emb1  = (const float*)d_in[3];
  const float* emb2  = (const float*)d_in[4];
  const float* W1    = (const float*)d_in[5];
  const float* b1    = (const float*)d_in[6];
  const float* W2    = (const float*)d_in[7];
  const float* b2    = (const float*)d_in[8];
  float* out = (float*)d_out;

  const int useWs = (d_ws != nullptr && ws_size >= (size_t)W1_BYTES) ? 1 : 0;
  if (useWs) {
    // one-time (per launch) W1 conversion + fragment swizzle into workspace
    fraud_prep_w1<<<dim3(16), dim3(THREADS), 0, stream>>>(
        W1, (unsigned int*)d_ws);
  }
  dim3 grid(B_TOTAL / ROWS_PB);   // 2048 blocks x 256 threads (8 wave32)
  fraud_mlp_wmma<<<grid, dim3(THREADS), 0, stream>>>(
      x_num, x_cat, emb0, emb1, emb2, W1, b1, W2, b2,
      (const unsigned int*)d_ws, useWs, out);
}
